// HyperEdgeAttention_85942295593116
// MI455X (gfx1250) — matrix operand order repaired and checked by reading, not compiled, not run on hardware
//
#include <hip/hip_runtime.h>

// B=16, N=4096, M=512, FE=FV=256, H=256, ND=128
#define Bb_ 16
#define Nn_ 4096
#define Mm_ 512
#define Hh_ 256
#define NDd_ 128

typedef __attribute__((ext_vector_type(16))) __bf16        v16bf;
typedef __attribute__((ext_vector_type(8)))  float         v8f;
typedef __attribute__((ext_vector_type(8)))  unsigned int  v8u;
typedef __attribute__((ext_vector_type(4)))  unsigned int  v4u;
typedef __attribute__((ext_vector_type(8)))  unsigned int  u32x8;
typedef __attribute__((ext_vector_type(4)))  unsigned int  u32x4;

__device__ __forceinline__ unsigned short f2bf(float f) {
  unsigned u = __builtin_bit_cast(unsigned, f);
  unsigned r = u + 0x7FFFu + ((u >> 16) & 1u);   // round-to-nearest-even
  return (unsigned short)(r >> 16);
}

// pack two floats -> packed bf16 dword (low = x0, high = x1)
__device__ __forceinline__ unsigned packbf(float x0, float x1) {
#if __has_builtin(__builtin_amdgcn_cvt_pk_bf16_f32)
  typedef __attribute__((ext_vector_type(2))) __bf16 v2bf;
  v2bf p = __builtin_amdgcn_cvt_pk_bf16_f32(x0, x1);
  return __builtin_bit_cast(unsigned, p);
#else
  return (unsigned)f2bf(x0) | ((unsigned)f2bf(x1) << 16);
#endif
}

// ---------------- small setup kernels ----------------

__global__ void combine_att(const float* __restrict__ W1, const float* __restrict__ W2,
                            const float* __restrict__ att,
                            float* __restrict__ w1v, float* __restrict__ w2v) {
  int f = threadIdx.x;
  float s = 0.f;
  if (blockIdx.x == 0) {
    for (int h = 0; h < Hh_; ++h) s += att[h] * W1[h * 256 + f];
    w1v[f] = s;
  } else {
    for (int h = 0; h < Hh_; ++h) s += att[Hh_ + h] * W2[h * 256 + f];
    w2v[f] = s;
  }
}

// out[row] = dot(X[row, 0:256], w)   (one wave32 per row, 8 rows/block)
__global__ __launch_bounds__(256) void rowdot256(const float* __restrict__ X,
                                                 const float* __restrict__ w,
                                                 float* __restrict__ out) {
  int lane = threadIdx.x & 31;
  int row  = blockIdx.x * 8 + (threadIdx.x >> 5);
  const float* x = X + (long long)row * 256;
  float s = 0.f;
#pragma unroll
  for (int i = 0; i < 8; ++i) s += x[lane + i * 32] * w[lane + i * 32];
#pragma unroll
  for (int off = 16; off; off >>= 1) s += __shfl_down(s, off, 32);
  if (lane == 0) out[row] = s;
}

// ---------------- masked column softmax (over n), streaming chunk stats --------

__device__ __forceinline__ float leaky01(float l) {
  l = l > 0.f ? l : 0.2f * l;
  return l * 0.01f;
}

__global__ __launch_bounds__(512) void col_chunk_stats(const float* __restrict__ IHG,
                                                       const float* __restrict__ e_att,
                                                       const float* __restrict__ v_att,
                                                       float* __restrict__ pmax,
                                                       float* __restrict__ psum) {
  const int m = threadIdx.x, chunk = blockIdx.x, b = blockIdx.y;
  const float ea = e_att[b * Mm_ + m];
  float mx = -__builtin_inff(), s = 0.f;
  const int n0 = chunk * 128;
  for (int j = 0; j < 128; ++j) {
    const int n = n0 + j;
    float I = IHG[(((long long)b * Nn_) + n) * Mm_ + m];
    if (I != 0.f) {
      float l = leaky01(v_att[b * Nn_ + n] + ea);
      if (l > mx) { s = s * __expf(mx - l) + 1.f; mx = l; }
      else        { s += __expf(l - mx); }
    }
  }
  const long long o = ((long long)(b * 32 + chunk)) * Mm_ + m;
  pmax[o] = mx; psum[o] = s;
}

__global__ __launch_bounds__(512) void col_merge(const float* __restrict__ pmax,
                                                 const float* __restrict__ psum,
                                                 float* __restrict__ colmax,
                                                 float* __restrict__ colinv) {
  const int m = threadIdx.x, b = blockIdx.x;
  float gm = -__builtin_inff();
  for (int c = 0; c < 32; ++c) gm = fmaxf(gm, pmax[(long long)(b * 32 + c) * Mm_ + m]);
  float s = 0.f;
  if (gm > -__builtin_inff()) {
    for (int c = 0; c < 32; ++c) {
      float pm = pmax[(long long)(b * 32 + c) * Mm_ + m];
      if (pm > -__builtin_inff())
        s += psum[(long long)(b * 32 + c) * Mm_ + m] * __expf(pm - gm);
    }
  }
  colmax[b * Mm_ + m] = gm;
  colinv[b * Mm_ + m] = s > 0.f ? 1.f / s : 0.f;   // empty hyperedge -> 0 (nan_to_num)
}

__global__ __launch_bounds__(256) void build_alpha(const float* __restrict__ IHG,
                                                   const float* __restrict__ e_att,
                                                   const float* __restrict__ v_att,
                                                   const float* __restrict__ colmax,
                                                   const float* __restrict__ colinv,
                                                   unsigned short* __restrict__ alpha) {
  long long idx = (long long)blockIdx.x * 256 + threadIdx.x;   // [b][n][m]
  int m = (int)(idx & (Mm_ - 1));
  long long bn = idx >> 9;                                     // b*N + n
  int b = (int)(idx >> 21);                                    // N*M = 2^21
  int bm = b * Mm_ + m;
  float I = IHG[idx];
  float a = 0.f, cv = colinv[bm];
  if (I != 0.f && cv > 0.f) {
    float l = leaky01(v_att[bn] + e_att[bm]);
    a = __expf(l - colmax[bm]) * cv;
  }
  alpha[idx] = f2bf(a);
}

// ---------------- fused bf16 WMMA GEMM (64x64 block tile, 8 waves) ----------------

enum { A_F32 = 0, A_F32_AFFRELU = 1, A_BF16_TDM = 2, A_IHG_T = 3 };
enum { B_KMAJ = 0, B_CMAJ = 1, B_KMAJ_AFF = 2 };

template <int AM, int BM, bool BIAS>
__global__ __launch_bounds__(256) void gemm_bf16_64x64(
    const void* __restrict__ Ap, const float* __restrict__ Bp, float* __restrict__ Cp,
    const float* __restrict__ ascale, const float* __restrict__ ashift,
    const float* __restrict__ bscale, const float* __restrict__ bshift,
    const float* __restrict__ bias,
    int K, int lda, int ldb, int ldc,
    long long aStride, long long bStride, long long cStride) {
  __shared__ __attribute__((aligned(32))) unsigned short As[64 * 32];  // [r][k]
  __shared__ __attribute__((aligned(32))) unsigned short Bt[64 * 32];  // [c][k] (transposed)
  const int tid = threadIdx.x;
  const int bz  = blockIdx.z;
  const float*          Af = (const float*)Ap + bz * aStride;
  const unsigned short* Ah = (const unsigned short*)Ap + bz * aStride;
  const float*          Bg = Bp + bz * bStride;
  float*                Cg = Cp + bz * cStride;
  const int r0 = blockIdx.y * 64, c0 = blockIdx.x * 64;

  const int w = tid >> 5, lane = tid & 31;
  const int wr = (w & 3) * 16, wc = (w >> 2) * 32;
  const int fm = lane & 15, kh = lane >> 4;

  v8f acc0 = {}, acc1 = {};

  for (int kt = 0; kt < K; kt += 32) {
    // ---- stage A tile 64x32 -> bf16 LDS ----
    if constexpr (AM == A_BF16_TDM) {
      // Tensor Data Mover: each wave DMAs its 8-row slice of the bf16 A tile.
      // D# per CDNA5 ISA: group0 = {count, lds_addr, gaddr, type=2},
      // group1 = {data_size=2B, tensor_dim0=lda, dim0_stride=lda, tile 32x8}.
      unsigned ldsA = (unsigned)(unsigned long long)(const void*)As
                    + (unsigned)(w * 8 * 32 * 2);
      unsigned long long ga =
          (unsigned long long)(const void*)(Ah + (long long)(r0 + w * 8) * lda + kt);
      u32x4 g0;
      g0[0] = 1u;                                                 // count=1 (valid D#)
      g0[1] = ldsA;                                               // LDS byte address
      g0[2] = (unsigned)ga;                                       // global addr [31:0]
      g0[3] = (unsigned)((ga >> 32) & 0x01FFFFFFu) | 0x80000000u; // [56:32] | type=2
      u32x8 g1;
      g1[0] = 1u << 16;                                           // data_size = 2 bytes
      g1[1] = ((unsigned)lda & 0xFFFFu) << 16;                    // tensor_dim0 lo16
      g1[2] = ((unsigned)lda >> 16) & 0xFFFFu;                    // dim0 hi | dim1 lo (=0)
      g1[3] = 16u | (32u << 16);                                  // dim1 hi (1<<20) | tile_dim0=32
      g1[4] = 8u;                                                 // tile_dim1=8, tile_dim2=0
      g1[5] = (unsigned)lda;                                      // tensor_dim0_stride lo32
      g1[6] = 0u;                                                 // stride hi | dim1_stride lo
      g1[7] = 0u;
      asm volatile("tensor_load_to_lds %0, %1, null, null"
                   :: "s"(g0), "s"(g1) : "memory");
    } else {
#pragma unroll
      for (int i = 0; i < 4; ++i) {           // 1024 bf16-pairs, packed b32 LDS stores
        int idx = tid + i * 256;
        int r, kp;
        if constexpr (AM == A_IHG_T) { r = idx & 63; kp = idx >> 6; }  // coalesced over r
        else                         { kp = idx & 15; r = idx >> 4; }  // coalesced over k
        const int k = kp * 2;
        float x0, x1;
        if constexpr (AM == A_IHG_T) {
          x0 = Af[(long long)(kt + k)     * lda + (r0 + r)];
          x1 = Af[(long long)(kt + k + 1) * lda + (r0 + r)];
        } else {
          const float* ap = &Af[(long long)(r0 + r) * lda + (kt + k)];
          x0 = ap[0]; x1 = ap[1];
          if constexpr (AM == A_F32_AFFRELU) {
            x0 = fmaxf(0.f, x0 * ascale[kt + k]     + ashift[kt + k]);
            x1 = fmaxf(0.f, x1 * ascale[kt + k + 1] + ashift[kt + k + 1]);
          }
        }
        *(unsigned*)&As[r * 32 + k] = packbf(x0, x1);
      }
    }
    // ---- stage B tile 32x64 -> transposed bf16 LDS [c][k], packed stores ----
#pragma unroll
    for (int i = 0; i < 4; ++i) {
      int idx = tid + i * 256;
      int kp, c;
      float x0, x1;
      if constexpr (BM == B_CMAJ) {                       // weights: W[c, k]
        kp = idx & 15; c = idx >> 4;
        const float* bp = &Bg[(long long)(c0 + c) * ldb + (kt + kp * 2)];
        x0 = bp[0]; x1 = bp[1];
      } else {
        c = idx & 63; kp = idx >> 6;
        x0 = Bg[(long long)(kt + kp * 2)     * ldb + (c0 + c)];
        x1 = Bg[(long long)(kt + kp * 2 + 1) * ldb + (c0 + c)];
        if constexpr (BM == B_KMAJ_AFF) {
          float sc = bscale[c0 + c], sh = bshift[c0 + c];
          x0 = x0 * sc + sh; x1 = x1 * sc + sh;
        }
      }
      *(unsigned*)&Bt[c * 32 + kp * 2] = packbf(x0, x1);
    }
    if constexpr (AM == A_BF16_TDM) __builtin_amdgcn_s_wait_tensorcnt(0);
    __syncthreads();
    // ---- per-lane fragments (contiguous k = 16*kh + s, same order in A and B) ----
    const unsigned short* ap = &As[(wr + fm) * 32 + kh * 16];
    v4u alo = *(const v4u*)ap;
    v4u ahi = *(const v4u*)(ap + 8);
    v8u afu;
    afu[0] = alo[0]; afu[1] = alo[1]; afu[2] = alo[2]; afu[3] = alo[3];
    afu[4] = ahi[0]; afu[5] = ahi[1]; afu[6] = ahi[2]; afu[7] = ahi[3];
    v16bf afr = __builtin_bit_cast(v16bf, afu);
    v8u b0u = *(const v8u*)&Bt[(wc + fm) * 32 + kh * 16];
    v8u b1u = *(const v8u*)&Bt[(wc + 16 + fm) * 32 + kh * 16];
    v16bf bf0 = __builtin_bit_cast(v16bf, b0u);
    v16bf bf1 = __builtin_bit_cast(v16bf, b1u);
    acc0 = __builtin_amdgcn_wmma_f32_16x16x32_bf16(false, afr, false, bf0, (short)0, acc0, false, false);
    acc1 = __builtin_amdgcn_wmma_f32_16x16x32_bf16(false, afr, false, bf1, (short)0, acc1, false, false);
    __syncthreads();
  }
  // ---- epilogue: C/D layout (VGPR r: lanes<16 -> M=r, lanes>=16 -> M=r+8) ----
  const int col = c0 + wc + fm;
  const int row = r0 + wr + (kh ? 8 : 0);
  float bi0 = 0.f, bi1 = 0.f;
  if constexpr (BIAS) { bi0 = bias[col]; bi1 = bias[col + 16]; }
#pragma unroll
  for (int r = 0; r < 8; ++r) {
    float o0 = acc0[r], o1 = acc1[r];
    if constexpr (BIAS) { o0 += bi0; o1 += bi1; }
    Cg[(long long)(row + r) * ldc + col]      = o0;
    Cg[(long long)(row + r) * ldc + col + 16] = o1;
  }
}

// ---------------- BatchNorm helpers (training-mode batch stats) ----------------

__global__ void zero_f32(float* __restrict__ p, int n) {
  int i = blockIdx.x * 256 + threadIdx.x;
  if (i < n) p[i] = 0.f;
}

template <int C>
__global__ void bn_stats(const float* __restrict__ X, float* __restrict__ sums,
                         float* __restrict__ sumsq) {
  int c = threadIdx.x;
  long long r0 = (long long)blockIdx.x * 256;
  float s = 0.f, s2 = 0.f;
  for (int r = 0; r < 256; ++r) {
    float x = X[(r0 + r) * C + c];
    s += x; s2 += x * x;
  }
  atomicAdd(&sums[c], s);
  atomicAdd(&sumsq[c], s2);
}

__global__ void bn_finalize(const float* __restrict__ sums, const float* __restrict__ sumsq,
                            const float* __restrict__ g, const float* __restrict__ beta,
                            float* __restrict__ scale, float* __restrict__ shift, float invR) {
  int c = threadIdx.x;
  float mean = sums[c] * invR;
  float var  = sumsq[c] * invR - mean * mean;     // biased variance
  float sc   = g[c] * rsqrtf(var + 1e-5f);
  scale[c] = sc;
  shift[c] = beta[c] - mean * sc;
}

__global__ void bn_apply_inplace(float* __restrict__ X, const float* __restrict__ scale,
                                 const float* __restrict__ shift, int Cmask) {
  long long i = (long long)blockIdx.x * 256 + threadIdx.x;
  int c = (int)(i & Cmask);
  X[i] = X[i] * scale[c] + shift[c];
}

// ---------------- orchestration ----------------

extern "C" void kernel_launch(void* const* d_in, const int* in_sizes, int n_in,
                              void* d_out, int out_size, void* d_ws, size_t ws_size,
                              hipStream_t stream) {
  const float* e_HG  = (const float*)d_in[0];
  const float* v_CG  = (const float*)d_in[1];
  const float* I_HG  = (const float*)d_in[2];
  const float* W1    = (const float*)d_in[3];
  const float* W2    = (const float*)d_in[4];
  const float* att   = (const float*)d_in[5];
  const float* fv_w1 = (const float*)d_in[6];
  const float* fv_b1 = (const float*)d_in[7];
  const float* fv_g1 = (const float*)d_in[8];
  const float* fv_bb1= (const float*)d_in[9];
  const float* fv_w2 = (const float*)d_in[10];
  const float* fv_b2 = (const float*)d_in[11];
  const float* fv_g2 = (const float*)d_in[12];
  const float* fv_bb2= (const float*)d_in[13];
  const float* f2_w1 = (const float*)d_in[14];
  const float* f2_b1 = (const float*)d_in[15];
  const float* f2_g1 = (const float*)d_in[16];
  const float* f2_bb1= (const float*)d_in[17];
  const float* f2_w2 = (const float*)d_in[18];
  const float* f2_b2 = (const float*)d_in[19];
  const float* f2_g2 = (const float*)d_in[20];
  const float* f2_bb2= (const float*)d_in[21];

  float* ws = (float*)d_ws;
  float* w1v    = ws;            float* w2v    = ws + 256;
  float* e_att  = ws + 512;      float* v_att  = ws + 8704;
  float* colmax = ws + 74240;    float* colinv = ws + 82432;
  float* pmax   = ws + 90624;    float* psum   = ws + 352768;
  float* sums   = ws + 614912;   float* sumsq  = ws + 615168;
  float* sc1 = ws + 615424, *sh1 = ws + 615680;
  float* sc2 = ws + 615936, *sh2 = ws + 616192;
  float* sc3 = ws + 616448, *sh3 = ws + 616704;
  float* sc4 = ws + 616960, *sh4 = ws + 617216;
  float* buf0 = ws + 622592;                 // 16,777,216 floats
  float* buf1 = buf0 + 16777216;             // 16,777,216 floats
  unsigned short* alpha = (unsigned short*)buf0;   // [B,N,M] bf16
  float* v1    = buf1;                       // [B*N, FE]
  float* x1pre = buf0;                       // [B*N, H]   (alpha dead after G1)
  float* x2pre = buf1;                       // [B*N, ND]  (v1 dead after G2)
  float* e2    = buf0;                       // [B*M, ND]  (x1pre dead after G3)
  float* y1pre = buf0 + 4194304;             // [B*M, H]
  float* out   = (float*)d_out;              // [B*M, ND]

  combine_att<<<2, 256, 0, stream>>>(W1, W2, att, w1v, w2v);
  rowdot256<<<(Bb_ * Mm_) / 8, 256, 0, stream>>>(e_HG, w1v, e_att);
  rowdot256<<<(Bb_ * Nn_) / 8, 256, 0, stream>>>(v_CG, w2v, v_att);
  col_chunk_stats<<<dim3(32, Bb_), 512, 0, stream>>>(I_HG, e_att, v_att, pmax, psum);
  col_merge<<<Bb_, 512, 0, stream>>>(pmax, psum, colmax, colinv);
  build_alpha<<<(Bb_ * (long long)Nn_ * Mm_) / 256, 256, 0, stream>>>(
      I_HG, e_att, v_att, colmax, colinv, alpha);

  // G1: v1 = alpha @ e_HG (per batch [4096x512]x[512x256]); A tile via TDM
  gemm_bf16_64x64<A_BF16_TDM, B_KMAJ, false><<<dim3(4, 64, Bb_), 256, 0, stream>>>(
      alpha, e_HG, v1, nullptr, nullptr, nullptr, nullptr, nullptr,
      512, 512, 256, 256, (long long)Nn_ * Mm_, (long long)Mm_ * 256, (long long)Nn_ * 256);

  // G2: x1pre = v1 @ fv_w1^T + fv_b1   [65536x256]x[256x256]
  gemm_bf16_64x64<A_F32, B_CMAJ, true><<<dim3(4, 1024, 1), 256, 0, stream>>>(
      v1, fv_w1, x1pre, nullptr, nullptr, nullptr, nullptr, fv_b1,
      256, 256, 256, 256, 0, 0, 0);
  zero_f32<<<2, 256, 0, stream>>>(sums, 512);
  bn_stats<256><<<256, 256, 0, stream>>>(x1pre, sums, sumsq);
  bn_finalize<<<1, 256, 0, stream>>>(sums, sumsq, fv_g1, fv_bb1, sc1, sh1, 1.f / 65536.f);

  // G3: x2pre = relu(bn(x1pre)) @ fv_w2^T + fv_b2   [65536x256]x[256x128]
  gemm_bf16_64x64<A_F32_AFFRELU, B_CMAJ, true><<<dim3(2, 1024, 1), 256, 0, stream>>>(
      x1pre, fv_w2, x2pre, sc1, sh1, nullptr, nullptr, fv_b2,
      256, 256, 256, 128, 0, 0, 0);
  zero_f32<<<2, 256, 0, stream>>>(sums, 512);
  bn_stats<128><<<256, 128, 0, stream>>>(x2pre, sums, sumsq);
  bn_finalize<<<1, 128, 0, stream>>>(sums, sumsq, fv_g2, fv_bb2, sc2, sh2, 1.f / 65536.f);

  // G4: e2 = I_HG^T @ bn(x2pre)   (per batch [512x4096]x[4096x128])
  gemm_bf16_64x64<A_IHG_T, B_KMAJ_AFF, false><<<dim3(2, 8, Bb_), 256, 0, stream>>>(
      I_HG, x2pre, e2, nullptr, nullptr, sc2, sh2, nullptr,
      4096, 512, 128, 128, (long long)Nn_ * Mm_, (long long)Nn_ * NDd_, (long long)Mm_ * NDd_);

  // G5: y1pre = e2 @ f2_w1^T + f2_b1   [8192x128]x[128x256]
  gemm_bf16_64x64<A_F32, B_CMAJ, true><<<dim3(4, 128, 1), 256, 0, stream>>>(
      e2, f2_w1, y1pre, nullptr, nullptr, nullptr, nullptr, f2_b1,
      128, 128, 128, 256, 0, 0, 0);
  zero_f32<<<2, 256, 0, stream>>>(sums, 512);
  bn_stats<256><<<32, 256, 0, stream>>>(y1pre, sums, sumsq);
  bn_finalize<<<1, 256, 0, stream>>>(sums, sumsq, f2_g1, f2_bb1, sc3, sh3, 1.f / 8192.f);

  // G6: out_pre = relu(bn(y1pre)) @ f2_w2^T + f2_b2   [8192x256]x[256x128]
  gemm_bf16_64x64<A_F32_AFFRELU, B_CMAJ, true><<<dim3(2, 128, 1), 256, 0, stream>>>(
      y1pre, f2_w2, out, sc3, sh3, nullptr, nullptr, f2_b2,
      256, 256, 256, 128, 0, 0, 0);
  zero_f32<<<2, 256, 0, stream>>>(sums, 512);
  bn_stats<128><<<32, 128, 0, stream>>>(out, sums, sumsq);
  bn_finalize<<<1, 128, 0, stream>>>(sums, sumsq, f2_g2, f2_bb2, sc4, sh4, 1.f / 8192.f);
  bn_apply_inplace<<<(Bb_ * Mm_ * NDd_) / 256, 256, 0, stream>>>(out, sc4, sh4, NDd_ - 1);
}
